// mh_attention_73203422593030
// MI455X (gfx1250) — compile-verified, matrix operand order checked by reading, not run on hardware
//
#include <hip/hip_runtime.h>

#define WAVES 8
#define CCH 128
#define WSTR 136   // bf16 weight row stride (padded, bank-conflict-free)
#define FSTR 132   // f32 staging row stride (padded)

typedef __attribute__((ext_vector_type(16))) __bf16 v16bf;
typedef __attribute__((ext_vector_type(8)))  __bf16 v8bf;
typedef __attribute__((ext_vector_type(8)))  float  v8f;
typedef __attribute__((ext_vector_type(4)))  float  fv4;   // native vec4 (NT-capable)

static __device__ inline fv4 ldg_nt(const float* p) {
  return __builtin_nontemporal_load((const fv4*)p);
}

#define WMMA_BF16(A, B, C) \
  __builtin_amdgcn_wmma_f32_16x16x32_bf16(false, (A), false, (B), (short)0, (C), false, false)

// Pack 16 floats (two 8-wide chunks per the 16-bit A/B WMMA lane layout) into v16bf
static __device__ inline v16bf cvt16(const fv4 c0, const fv4 c1,
                                     const fv4 c2, const fv4 c3) {
  v16bf r;
  r[0]=(__bf16)c0.x;  r[1]=(__bf16)c0.y;  r[2]=(__bf16)c0.z;  r[3]=(__bf16)c0.w;
  r[4]=(__bf16)c1.x;  r[5]=(__bf16)c1.y;  r[6]=(__bf16)c1.z;  r[7]=(__bf16)c1.w;
  r[8]=(__bf16)c2.x;  r[9]=(__bf16)c2.y;  r[10]=(__bf16)c2.z; r[11]=(__bf16)c2.w;
  r[12]=(__bf16)c3.x; r[13]=(__bf16)c3.y; r[14]=(__bf16)c3.z; r[15]=(__bf16)c3.w;
  return r;
}

// A fragment from one 128-float row: lane holds K = {ks*32+off+0..7, ks*32+off+16..23}
static __device__ inline v16bf cvt_fragA(const float* __restrict__ rowp, int ks, int off) {
  return cvt16(ldg_nt(rowp + ks * 32 + off),      ldg_nt(rowp + ks * 32 + off + 4),
               ldg_nt(rowp + ks * 32 + off + 16), ldg_nt(rowp + ks * 32 + off + 20));
}

// Same, but element-wise sum of two rows (new_feature + xyz_enc)
static __device__ inline v16bf cvt_fragA2(const float* __restrict__ r1,
                                          const float* __restrict__ r2, int ks, int off) {
  fv4 c0 = ldg_nt(r1 + ks * 32 + off)      + ldg_nt(r2 + ks * 32 + off);
  fv4 c1 = ldg_nt(r1 + ks * 32 + off + 4)  + ldg_nt(r2 + ks * 32 + off + 4);
  fv4 c2 = ldg_nt(r1 + ks * 32 + off + 16) + ldg_nt(r2 + ks * 32 + off + 16);
  fv4 c3 = ldg_nt(r1 + ks * 32 + off + 20) + ldg_nt(r2 + ks * 32 + off + 20);
  return cvt16(c0, c1, c2, c3);
}

// B fragment (transposed weight): lane col j = t*16+col, rows i = ks*32+off+{0..7,16..23}
static __device__ inline v16bf load_bfrag(const __bf16* __restrict__ w, int t, int ks,
                                          int col, int off) {
  const __bf16* base = w + (t * 16 + col) * WSTR + ks * 32 + off;
  v8bf lo = *(const v8bf*)(base);
  v8bf hi = *(const v8bf*)(base + 16);
  return __builtin_shufflevector(lo, hi, 0,1,2,3,4,5,6,7,8,9,10,11,12,13,14,15);
}

// Single-A projection (Q): pipelined B-fragment loads (distinct regs, next tile prefetched)
static __device__ inline void proj_store(const v16bf* __restrict__ afrag,
                                         const __bf16* __restrict__ w,
                                         const float* __restrict__ bias_lds,
                                         float* __restrict__ dst,
                                         int col, int hi, int off) {
  v16bf b0 = load_bfrag(w, 0, 0, col, off);
  v16bf b1 = load_bfrag(w, 0, 1, col, off);
  v16bf b2 = load_bfrag(w, 0, 2, col, off);
  v16bf b3 = load_bfrag(w, 0, 3, col, off);
#pragma unroll
  for (int t = 0; t < 8; ++t) {
    v16bf n0 = b0, n1 = b1, n2 = b2, n3 = b3;
    if (t < 7) {
      n0 = load_bfrag(w, t + 1, 0, col, off);
      n1 = load_bfrag(w, t + 1, 1, col, off);
      n2 = load_bfrag(w, t + 1, 2, col, off);
      n3 = load_bfrag(w, t + 1, 3, col, off);
    }
    v8f acc = {};
    acc = WMMA_BF16(afrag[0], b0, acc);
    acc = WMMA_BF16(afrag[1], b1, acc);
    acc = WMMA_BF16(afrag[2], b2, acc);
    acc = WMMA_BF16(afrag[3], b3, acc);
    float bv = bias_lds[t * 16 + col];
#pragma unroll
    for (int r = 0; r < 8; ++r)
      dst[(r + hi * 8) * FSTR + t * 16 + col] = acc[r] + bv;
    b0 = n0; b1 = n1; b2 = n2; b3 = n3;
  }
}

// Dual-A projection: one B-fragment load feeds TWO WMMAs; next tile double-buffered
static __device__ inline void proj2_store(const v16bf* __restrict__ a0,
                                          const v16bf* __restrict__ a1,
                                          const __bf16* __restrict__ w,
                                          const float* __restrict__ bias_lds,
                                          float* __restrict__ dst0,
                                          float* __restrict__ dst1,
                                          int col, int hi, int off) {
  v16bf b0 = load_bfrag(w, 0, 0, col, off);
  v16bf b1 = load_bfrag(w, 0, 1, col, off);
  v16bf b2 = load_bfrag(w, 0, 2, col, off);
  v16bf b3 = load_bfrag(w, 0, 3, col, off);
#pragma unroll
  for (int t = 0; t < 8; ++t) {
    v16bf n0 = b0, n1 = b1, n2 = b2, n3 = b3;
    if (t < 7) {
      n0 = load_bfrag(w, t + 1, 0, col, off);
      n1 = load_bfrag(w, t + 1, 1, col, off);
      n2 = load_bfrag(w, t + 1, 2, col, off);
      n3 = load_bfrag(w, t + 1, 3, col, off);
    }
    v8f acc0 = {};
    v8f acc1 = {};
    acc0 = WMMA_BF16(a0[0], b0, acc0);
    acc1 = WMMA_BF16(a1[0], b0, acc1);
    acc0 = WMMA_BF16(a0[1], b1, acc0);
    acc1 = WMMA_BF16(a1[1], b1, acc1);
    acc0 = WMMA_BF16(a0[2], b2, acc0);
    acc1 = WMMA_BF16(a1[2], b2, acc1);
    acc0 = WMMA_BF16(a0[3], b3, acc0);
    acc1 = WMMA_BF16(a1[3], b3, acc1);
    float bv = bias_lds[t * 16 + col];
#pragma unroll
    for (int r = 0; r < 8; ++r) {
      dst0[(r + hi * 8) * FSTR + t * 16 + col] = acc0[r] + bv;
      dst1[(r + hi * 8) * FSTR + t * 16 + col] = acc1[r] + bv;
    }
    b0 = n0; b1 = n1; b2 = n2; b3 = n3;
  }
}

__global__ __launch_bounds__(WAVES * 32, 1) void pt_attn_kernel(
    const float* __restrict__ x_feat,      // [N,128]
    const float* __restrict__ xyz_enc,     // [N,16,128]
    const float* __restrict__ new_feature, // [N,16,128]
    const float* __restrict__ Wq, const float* __restrict__ bq,
    const float* __restrict__ Wk, const float* __restrict__ bk,
    const float* __restrict__ Wv, const float* __restrict__ bv,
    float* __restrict__ out, int n) {
  __shared__ __bf16 wlds[3 * CCH * WSTR];        // bf16 weights, padded rows (102 KB)
  __shared__ float  blds[3][CCH];                // biases (1.5 KB)
  __shared__ float  qbuf[WAVES][16 * FSTR];      // per-wave Q  (16 points x 128)  (66 KB)
  __shared__ float  kvbuf[WAVES][2][16 * FSTR];  // per-wave Xk/Xv for 2 points   (132 KB)
  __shared__ float  attb[WAVES][2][64];          // per-wave softmax weights       (4 KB)

  const int tid  = threadIdx.x;
  const int wave = tid >> 5;
  const int lane = tid & 31;
  const int col  = lane & 15;
  const int hi   = lane >> 4;
  const int off  = hi * 8;

  // Preload + convert weights (and biases) to LDS, once per block
  for (int idx = tid; idx < CCH * CCH; idx += WAVES * 32) {
    int r = idx >> 7, c = idx & 127;
    wlds[0 * CCH * WSTR + r * WSTR + c] = (__bf16)Wq[idx];
    wlds[1 * CCH * WSTR + r * WSTR + c] = (__bf16)Wk[idx];
    wlds[2 * CCH * WSTR + r * WSTR + c] = (__bf16)Wv[idx];
  }
  if (tid < CCH) {
    blds[0][tid] = bq[tid];
    blds[1][tid] = bk[tid];
    blds[2][tid] = bv[tid];
  }
  __syncthreads();

  const __bf16* wq = wlds;
  const __bf16* wk = wlds + CCH * WSTR;
  const __bf16* wv = wlds + 2 * CCH * WSTR;
  float* qb = qbuf[wave];

  const int groups = (n + 15) >> 4;
  v16bf afrag0[4];
  v16bf afrag1[4];

  for (int g = blockIdx.x * WAVES + wave; g < groups; g += gridDim.x * WAVES) {
    // ---- Q projection: M = 16 points of this group
    {
      int pq = g * 16 + col; if (pq >= n) pq = n - 1;
      const float* rowp = x_feat + (size_t)pq * CCH;
#pragma unroll
      for (int ks = 0; ks < 4; ++ks) afrag0[ks] = cvt_fragA(rowp, ks, off);
      proj_store(afrag0, wq, blds[0], qb, col, hi, off);
    }

    // ---- per pair of points of the group
    for (int p = 0; p < 16; p += 2) {
      const int pt0 = g * 16 + p;
      const int pt1 = pt0 + 1;
      const int pc0 = (pt0 < n) ? pt0 : (n - 1);
      const int pc1 = (pt1 < n) ? pt1 : (n - 1);
      {
        const float* e0 = xyz_enc     + ((size_t)pc0 * 16 + col) * CCH;
        const float* f0 = new_feature + ((size_t)pc0 * 16 + col) * CCH;
        const float* e1 = xyz_enc     + ((size_t)pc1 * 16 + col) * CCH;
        const float* f1 = new_feature + ((size_t)pc1 * 16 + col) * CCH;
#pragma unroll
        for (int ks = 0; ks < 4; ++ks) {
          afrag0[ks] = cvt_fragA2(e0, f0, ks, off);
          afrag1[ks] = cvt_fragA2(e1, f1, ks, off);
        }
      }

      // Prefetch next point-pair's neighbor rows (global_prefetch_b8) to hide HBM latency
      {
        int pn = pt0 + 2; if (pn >= n) pn = n - 1;
        const float* pe = xyz_enc     + ((size_t)pn * 16 + col) * CCH;
        const float* pf = new_feature + ((size_t)pn * 16 + col) * CCH;
#pragma unroll
        for (int ks = 0; ks < 4; ++ks) {
          __builtin_prefetch(pe + ks * 32, 0, 3);
          __builtin_prefetch(pf + ks * 32, 0, 3);
        }
      }

      // K projection for both points (B-fragments loaded once, double-buffered)
      proj2_store(afrag0, afrag1, wk, blds[1], kvbuf[wave][0], kvbuf[wave][1], col, hi, off);

      // attention scores + softmax for both points
      // reference reshape: row = h*4 + kk/4, chan = (kk%4)*32 + d
#pragma unroll
      for (int pp = 0; pp < 2; ++pp) {
        const float* kvb = kvbuf[wave][pp];
        float* ab = attb[wave][pp];
#pragma unroll
        for (int it = 0; it < 2; ++it) {
          const int pair = it * 32 + lane;
          const int h = pair >> 4, kk = pair & 15;
          const float* qrow = qb  + (p + pp) * FSTR + h * 32;
          const float* krow = kvb + (h * 4 + (kk >> 2)) * FSTR + (kk & 3) * 32;
          float s = 0.f;
#pragma unroll
          for (int d = 0; d < 32; d += 4) {
            fv4 qv = *(const fv4*)(qrow + d);
            fv4 kv = *(const fv4*)(krow + d);
            s += qv.x * kv.x + qv.y * kv.y + qv.z * kv.z + qv.w * kv.w;
          }
          s *= 0.08838834764831845f;  // 1/sqrt(128)
          float m = s;
#pragma unroll
          for (int msk = 8; msk >= 1; msk >>= 1)
            m = fmaxf(m, __shfl_xor(m, msk, 32));
          float e = __expf(s - m);
          float sum = e;
#pragma unroll
          for (int msk = 8; msk >= 1; msk >>= 1)
            sum += __shfl_xor(sum, msk, 32);
          ab[pair] = e / sum;
        }
      }

      // V projection for both points (overwrites kv tiles; same-wave DS ops in-order)
      proj2_store(afrag0, afrag1, wv, blds[2], kvbuf[wave][0], kvbuf[wave][1], col, hi, off);

      // out[pt, h*32+d] = sum_kk att[h,kk] * Xv[h*4+kk/4, (kk%4)*32+d]
      const int h2 = lane >> 3;
      const int d0 = (lane * 4) & 31;
#pragma unroll
      for (int pp = 0; pp < 2; ++pp) {
        const float* kvb = kvbuf[wave][pp];
        const float* abh = attb[wave][pp] + h2 * 16;
        fv4 o = {0.f, 0.f, 0.f, 0.f};
#pragma unroll
        for (int kk = 0; kk < 16; ++kk) {
          float a = abh[kk];
          fv4 vv = *(const fv4*)(kvb + (h2 * 4 + (kk >> 2)) * FSTR + (kk & 3) * 32 + d0);
          o += a * vv;
        }
        const int pt = pt0 + pp;
        if (pt < n) {
          __builtin_nontemporal_store(o, (fv4*)(out + (size_t)pt * CCH + lane * 4));
        }
      }
    }
  }
}

extern "C" void kernel_launch(void* const* d_in, const int* in_sizes, int n_in,
                              void* d_out, int out_size, void* d_ws, size_t ws_size,
                              hipStream_t stream) {
  const float* x_feat      = (const float*)d_in[0];
  const float* xyz_enc     = (const float*)d_in[1];
  const float* new_feature = (const float*)d_in[2];
  // d_in[3]=head, d_in[4]=d_k (compile-time constants 4 / 32)
  const float* Wq = (const float*)d_in[5];
  const float* bq = (const float*)d_in[6];
  const float* Wk = (const float*)d_in[7];
  const float* bk = (const float*)d_in[8];
  const float* Wv = (const float*)d_in[9];
  const float* bv = (const float*)d_in[10];
  float* out = (float*)d_out;

  const int n = in_sizes[0] / CCH;
  const int groups = (n + 15) / 16;
  int blocks = (groups + WAVES - 1) / WAVES;
  if (blocks < 1) blocks = 1;

  pt_attn_kernel<<<blocks, WAVES * 32, 0, stream>>>(
      x_feat, xyz_enc, new_feature, Wq, bq, Wk, bk, Wv, bv, out, n);
}
